// GPSLayer_7275674600089
// MI455X (gfx1250) — matrix operand order repaired and checked by reading, not compiled, not run on hardware
//
#include <hip/hip_runtime.h>
#include <hip/hip_bf16.h>

typedef unsigned short u16;
typedef __attribute__((ext_vector_type(16))) __bf16 v16bf;
typedef __attribute__((ext_vector_type(8)))  float  v8f;

union AB { uint4 u[2]; v16bf f; };

__device__ __forceinline__ u16 f2bf(float f) {
    unsigned u = __float_as_uint(f);
    unsigned r = u + 0x7fffu + ((u >> 16) & 1u);   // round-to-nearest-even
    return (u16)(r >> 16);
}

// ---------------- small elementwise kernels ----------------
__global__ void k_zero(float* __restrict__ d, int n) {
    int i = blockIdx.x * blockDim.x + threadIdx.x;
    if (i < n) d[i] = 0.0f;
}
__global__ void k_f32_bf16(const float* __restrict__ s, u16* __restrict__ d, int n) {
    int i = blockIdx.x * blockDim.x + threadIdx.x;
    if (i < n) d[i] = f2bf(s[i]);
}
__global__ void k_add_bf16(const float* __restrict__ a, const float* __restrict__ b,
                           u16* __restrict__ d, int n) {
    int i = blockIdx.x * blockDim.x + threadIdx.x;
    if (i < n) d[i] = f2bf(a[i] + b[i]);
}

// ---------------- GINE edge aggregation ----------------
// 128 threads per edge, D=384 -> 3 elements per thread; relu(x[src]+ea) atomically into agg[dst]
__global__ __launch_bounds__(256) void k_edge(const float* __restrict__ x,
                                              const int* __restrict__ ei,
                                              const float* __restrict__ ea,
                                              float* __restrict__ agg, int E, int D) {
    int t = blockIdx.x * blockDim.x + threadIdx.x;
    int e = t >> 7;
    int d0 = t & 127;
    if (e >= E) return;
    int src = ei[e];
    int dst = ei[E + e];
    const float* xs  = x   + (size_t)src * D;
    const float* eas = ea  + (size_t)e   * D;
    float*       ag  = agg + (size_t)dst * D;
#pragma unroll
    for (int j = 0; j < 3; ++j) {
        int d = d0 + j * 128;
        float v = xs[d] + eas[d];
        v = v > 0.0f ? v : 0.0f;
        atomicAdd(&ag[d], v);
    }
}

// ---------------- generic bf16 WMMA GEMM ----------------
// out = act( A[M,K]bf16 @ W[K,cols]bf16 + bias ) * scale (+ res) ; writes f32 and/or bf16
// block = 256 (8 waves); wave w -> rows [blockIdx.y*128 + 16w, +16), cols [blockIdx.x*16, +16)
// The block's K x 16 weight panel is staged ONCE, transposed+padded, into LDS (single barrier);
// the inner loop is then pure global b128 (A frag) + LDS b128 (B frag) + WMMA, no sync.
#define GEMM_KMAX 768
__global__ __launch_bounds__(256) void k_gemm(const u16* __restrict__ A,
                                              const u16* __restrict__ W,
                                              const float* __restrict__ bias,
                                              const float* __restrict__ res,
                                              float* __restrict__ outf,
                                              u16* __restrict__ outb,
                                              int M, int K, int cols, int relu, float scale) {
    __shared__ __align__(16) u16 ldsw[16 * (GEMM_KMAX + 8)];   // ldsw[n*Kpad + k]
    int tid  = threadIdx.x;
    int wave = tid >> 5, lane = tid & 31, half = lane >> 4, n16 = lane & 15;
    int c0   = blockIdx.x * 16;
    int row0 = blockIdx.y * 128 + wave * 16;
    int rowA = row0 + n16;
    int Kpad = K + 8;   // K%32==0 -> Kpad%8==0: keeps 16B alignment, kills bank conflicts

    // ---- stage transposed W panel (global-coalesced reads, once per block) ----
    for (int base = tid * 2; base < K * 16; base += 512) {
        int kk = base >> 4;
        int nn = base & 15;                 // even
        unsigned w2 = *(const unsigned*)(W + (size_t)kk * cols + c0 + nn);
        ldsw[nn * Kpad + kk]       = (u16)(w2 & 0xffffu);
        ldsw[(nn + 1) * Kpad + kk] = (u16)(w2 >> 16);
    }
    __syncthreads();

    v8f acc = {0.f, 0.f, 0.f, 0.f, 0.f, 0.f, 0.f, 0.f};
    const u16* lpan = ldsw + n16 * Kpad + 8 * half;    // B frag: K runs {8h..8h+7} and {16+8h..16+8h+7}

    for (int k0 = 0; k0 < K; k0 += 32) {
        // A fragment: lane(half) K runs {8h..8h+7} and {16+8h..16+8h+7}, contiguous 16B each
        AB aU, bU;
        const u16* ap = A + (size_t)rowA * K + k0 + 8 * half;
        aU.u[0] = *(const uint4*)ap;
        aU.u[1] = *(const uint4*)(ap + 16);
        if (k0 + 32 < K) __builtin_prefetch(ap + 32, 0, 1);   // global_prefetch_b8

        bU.u[0] = *(const uint4*)(lpan + k0);
        bU.u[1] = *(const uint4*)(lpan + k0 + 16);

        acc = __builtin_amdgcn_wmma_f32_16x16x32_bf16(false, aU.f, false, bU.f,
                                                      (short)0, acc, false, false);
    }

    int col = c0 + n16;
    float bcol = bias[col];
#pragma unroll
    for (int r = 0; r < 8; ++r) {
        int ro = row0 + r + 8 * half;
        float v = acc[r] + bcol;
        if (relu) v = v > 0.0f ? v : 0.0f;
        v *= scale;
        if (res)  v += res[(size_t)ro * cols + col];
        if (outf) outf[(size_t)ro * cols + col] = v;
        if (outb) outb[(size_t)ro * cols + col] = f2bf(v);
    }
}

// ---------------- flash attention (bf16 WMMA, f32 accum) ----------------
// One wave per 16-query tile per (b,h). DH=48 padded to 64 in the contraction.
// Ping-pong P buffers -> a single block barrier per 32-key step.
__global__ __launch_bounds__(256) void k_attn(const u16* __restrict__ qb,
                                              const u16* __restrict__ kb,
                                              const u16* __restrict__ vb,
                                              u16* __restrict__ ob) {
    constexpr int D = 384, DH = 48, S = 1024;
    __shared__ __align__(16) u16 ldsp[8 * 1024];   // per-wave two 16x32 bf16 P tiles

    int tid = threadIdx.x;
    int wave = tid >> 5, lane = tid & 31, half = lane >> 4, n16 = lane & 15;
    int wid = blockIdx.x * 8 + wave;
    int qt = wid & 63;
    int h  = (wid >> 6) & 7;
    int b  = wid >> 9;
    size_t rowbase = (size_t)b * S + (size_t)qt * 16;
    const __bf16* vbh = (const __bf16*)vb;

    // Q fragments for K-chunks [0,32) and [32,64) (zeros beyond 48)
    AB qa[2];
    {
        const u16* qp0 = qb + (rowbase + n16) * D + h * DH;
#pragma unroll
        for (int s = 0; s < 2; ++s) {
            const u16* qp = qp0 + 32 * s + 8 * half;
            qa[s].u[0] = *(const uint4*)qp;            // K = 32s+8h .. +7 (<48 always)
            if (s == 0) {
                qa[s].u[1] = *(const uint4*)(qp + 16); // K = 16+8h .. +7
            } else {
                uint4 z; z.x = z.y = z.z = z.w = 0;    // K >= 48 -> zero pad
                qa[s].u[1] = z;
            }
        }
    }

    v8f o0 = {0.f,0.f,0.f,0.f,0.f,0.f,0.f,0.f};
    v8f o1 = o0, o2 = o0;
    float m[8], l[8];
#pragma unroll
    for (int r = 0; r < 8; ++r) { m[r] = -3.0e38f; l[r] = 0.0f; }

    u16* lpbase = ldsp + wave * 1024;
    int pb = 0;

    for (int kv0 = 0; kv0 < S; kv0 += 32) {
        // ---- scores: two 16x16 tiles (key groups g=0,1), contraction over padded 64 ----
        v8f sc0, sc1;
#pragma unroll
        for (int g = 0; g < 2; ++g) {
            v8f sa = {0.f,0.f,0.f,0.f,0.f,0.f,0.f,0.f};
#pragma unroll
            for (int s = 0; s < 2; ++s) {
                AB bU;
                int kk0 = 32 * s + 8 * half;           // B: K = kk0..+7 and kk0+16..+7 (zero if >= 48)
                if (kk0 < DH) {
                    const u16* kp = kb + ((size_t)b * S + kv0 + g * 16 + n16) * D + h * DH + kk0;
                    bU.u[0] = *(const uint4*)kp;
                    if (s == 0) bU.u[1] = *(const uint4*)(kp + 16); else { uint4 z2; z2.x = z2.y = z2.z = z2.w = 0; bU.u[1] = z2; }
                } else {
                    uint4 z; z.x = z.y = z.z = z.w = 0;
                    bU.u[0] = z; bU.u[1] = z;
                }
                sa = __builtin_amdgcn_wmma_f32_16x16x32_bf16(false, qa[s].f, false, bU.f,
                                                             (short)0, sa, false, false);
            }
            if (g == 0) sc0 = sa; else sc1 = sa;
        }

        // ---- online softmax over 32 keys; stage P into ping-pong LDS buffer in bf16 ----
        u16* lp = lpbase + pb * 512;
#pragma unroll
        for (int r = 0; r < 8; ++r) {
            float v0 = sc0[r], v1 = sc1[r];
            float mr = fmaxf(v0, v1);
            mr = fmaxf(mr, __shfl_xor(mr, 1, 32));
            mr = fmaxf(mr, __shfl_xor(mr, 2, 32));
            mr = fmaxf(mr, __shfl_xor(mr, 4, 32));
            mr = fmaxf(mr, __shfl_xor(mr, 8, 32));     // reduce across 16-lane half
            float mn = fmaxf(m[r], mr);
            float cr = __expf(m[r] - mn);
            float p0 = __expf(v0 - mn);
            float p1 = __expf(v1 - mn);
            l[r] = l[r] * cr + p0 + p1;
            m[r] = mn;
            int rr = r + 8 * half;                     // C-frag row
            lp[rr * 32 + n16]      = f2bf(p0);
            lp[rr * 32 + 16 + n16] = f2bf(p1);
            o0[r] *= cr; o1[r] *= cr; o2[r] *= cr;
        }
        __syncthreads();   // order this wave's P writes before its cross-lane reads

        // ---- P (16x32) as A fragment from LDS: two contiguous 16B reads per lane ----
        AB pU;
        const u16* pp = lp + n16 * 32 + 8 * half;
        pU.u[0] = *(const uint4*)pp;
        pU.u[1] = *(const uint4*)(pp + 16);
        pb ^= 1;

        // ---- o += P @ V for the 3 column tiles of DH=48 ----
#pragma unroll
        for (int t = 0; t < 3; ++t) {
            v16bf vB;
            const __bf16* vp = vbh + ((size_t)b * S + kv0 + 8 * half) * D + h * DH + t * 16 + n16;
#pragma unroll
            for (int e = 0; e < 16; ++e) vB[e] = vp[(size_t)(e + (e >> 3) * 8) * D];  // key = kv0 + 8*half + e (+8 for e>=8)
            if (t == 0)      o0 = __builtin_amdgcn_wmma_f32_16x16x32_bf16(false, pU.f, false, vB, (short)0, o0, false, false);
            else if (t == 1) o1 = __builtin_amdgcn_wmma_f32_16x16x32_bf16(false, pU.f, false, vB, (short)0, o1, false, false);
            else             o2 = __builtin_amdgcn_wmma_f32_16x16x32_bf16(false, pU.f, false, vB, (short)0, o2, false, false);
        }
    }

    // ---- finalize: divide by row sums, store bf16 ----
#pragma unroll
    for (int r = 0; r < 8; ++r) {
        float lr = l[r];
        lr += __shfl_xor(lr, 1, 32);
        lr += __shfl_xor(lr, 2, 32);
        lr += __shfl_xor(lr, 4, 32);
        lr += __shfl_xor(lr, 8, 32);
        float inv = 1.0f / lr;
        size_t rg = rowbase + r + 8 * half;
        ob[rg * D + h * DH + n16]      = f2bf(o0[r] * inv);
        ob[rg * D + h * DH + 16 + n16] = f2bf(o1[r] * inv);
        ob[rg * D + h * DH + 32 + n16] = f2bf(o2[r] * inv);
    }
}

// ---------------- BatchNorm (coalesced column sums + finalize) ----------------
// blockDim = 384 (one column per thread), each block sweeps `rowsPerBlock` rows coalesced,
// then two global f32 atomic adds per thread into sums[0..cols)=sum, [cols..2cols)=sumsq.
__global__ void k_col_sums(const float* __restrict__ x, float* __restrict__ sums,
                           int rowsPerBlock, int cols) {
    int col = threadIdx.x;
    int r0 = blockIdx.x * rowsPerBlock;
    float s = 0.f, ss = 0.f;
    for (int r = 0; r < rowsPerBlock; ++r) {
        float v = x[(size_t)(r0 + r) * cols + col];
        s += v; ss += v * v;
    }
    atomicAdd(&sums[col], s);
    atomicAdd(&sums[cols + col], ss);
}
__global__ void k_stats_fin(const float* __restrict__ sums, float* __restrict__ st,
                            int rows, int cols) {
    int c = blockIdx.x * blockDim.x + threadIdx.x;
    if (c >= cols) return;
    float mn = sums[c] / (float)rows;
    float var = sums[cols + c] / (float)rows - mn * mn;
    st[c] = mn;
    st[cols + c] = rsqrtf(var + 1e-5f);
}

__global__ void k_combine(const float* __restrict__ hl, const float* __restrict__ ha,
                          const float* __restrict__ stl, const float* __restrict__ sta,
                          const float* __restrict__ g1l, const float* __restrict__ be1l,
                          const float* __restrict__ g1a, const float* __restrict__ be1a,
                          float* __restrict__ h, u16* __restrict__ hb, int n, int D) {
    int i = blockIdx.x * blockDim.x + threadIdx.x;
    if (i >= n) return;
    int d = i % D;
    float v = (hl[i] - stl[d]) * stl[D + d] * g1l[d] + be1l[d]
            + (ha[i] - sta[d]) * sta[D + d] * g1a[d] + be1a[d];
    h[i] = v;
    hb[i] = f2bf(v);
}

__global__ void k_bn_out(const float* __restrict__ f2, const float* __restrict__ st,
                         const float* __restrict__ g, const float* __restrict__ be,
                         float* __restrict__ out, int n, int D) {
    int i = blockIdx.x * blockDim.x + threadIdx.x;
    if (i >= n) return;
    int d = i % D;
    out[i] = (f2[i] - st[d]) * st[D + d] * g[d] + be[d];
}

// ---------------- launch ----------------
extern "C" void kernel_launch(void* const* d_in, const int* in_sizes, int n_in,
                              void* d_out, int out_size, void* d_ws, size_t ws_size,
                              hipStream_t stream) {
    (void)in_sizes; (void)n_in; (void)out_size; (void)ws_size;
    constexpr int N = 32768, D = 384, E = 524288, FF = 768, S = 1024, Hh = 8;
    constexpr int ND = N * D;
    constexpr size_t A4 = (size_t)N * D * 4;   // f32 [N,D]
    constexpr size_t H2 = (size_t)N * D * 2;   // bf16 [N,D]

    const float* x   = (const float*)d_in[0];
    const int*   ei  = (const int*)d_in[1];
    const float* ea  = (const float*)d_in[2];
    const float* W1  = (const float*)d_in[4];  const float* b1  = (const float*)d_in[5];
    const float* W2  = (const float*)d_in[6];  const float* b2  = (const float*)d_in[7];
    const float* g1l = (const float*)d_in[8];  const float* be1l= (const float*)d_in[9];
    const float* Wq  = (const float*)d_in[10]; const float* bq  = (const float*)d_in[11];
    const float* Wk  = (const float*)d_in[12]; const float* bk  = (const float*)d_in[13];
    const float* Wv  = (const float*)d_in[14]; const float* bv  = (const float*)d_in[15];
    const float* Wo  = (const float*)d_in[16]; const float* bo  = (const float*)d_in[17];
    const float* g1a = (const float*)d_in[18]; const float* be1a= (const float*)d_in[19];
    const float* Wf1 = (const float*)d_in[20]; const float* bf1 = (const float*)d_in[21];
    const float* Wf2 = (const float*)d_in[22]; const float* bf2 = (const float*)d_in[23];
    const float* g2  = (const float*)d_in[24]; const float* be2 = (const float*)d_in[25];

    unsigned char* ws = (unsigned char*)d_ws;
    // workspace layout (phase-aliased)
    u16*   xb   = (u16*)(ws);                             // x bf16, whole run
    u16*   wgt  = (u16*)(ws + H2);                        // packed bf16 weights (2949120 B)
    u16*   W1b  = wgt;
    u16*   W2b  = wgt + D * D;
    u16*   Wqb  = wgt + 2 * D * D;
    u16*   Wkb  = wgt + 3 * D * D;
    u16*   Wvb  = wgt + 4 * D * D;
    u16*   Wob  = wgt + 5 * D * D;
    u16*   Wf1b = wgt + 6 * D * D;
    u16*   Wf2b = wgt + 6 * D * D + D * FF;
    float* st_l = (float*)(ws + H2 + 2949120);            // 3 stat buffers + 1 sums buffer
    float* st_a = st_l + 2 * D;
    float* st_2 = st_l + 4 * D;
    float* sums = st_l + 6 * D;                           // 2*D f32 scratch for column sums
    unsigned char* p = ws + H2 + 2949120 + 12288;
    float* agg  = (float*)p;                              // agg -> hl -> f2       (A4)
    float* hl   = agg;
    float* f2b_ = agg;
    u16*   hinb = (u16*)(p + A4);                         // hin_bf -> k_bf        (H2)
    u16*   kbf  = hinb;
    u16*   h1b  = (u16*)(p + A4 + H2);                    // h1_bf -> q_bf         (H2)
    u16*   qbf  = h1b;
    u16*   vbf  = (u16*)(p + A4 + 2 * H2);                // v_bf                  (H2)
    u16*   obf  = (u16*)(p + A4 + 3 * H2);                // o_bf -> h_bf          (H2)
    u16*   hbf  = obf;
    float* ha   = (float*)(p + A4 + 4 * H2);              // ha -> f1_bf           (A4)
    u16*   f1b  = (u16*)ha;
    float* hcomb= (float*)(p + 2 * A4 + 4 * H2);          // h f32                 (A4)

    const int T = 256;
    const int RPB = 256;                                  // rows per k_col_sums block
    const float qscale = 0.144337567297406441f;           // 1/sqrt(48)

    // pack activations & weights to bf16
    k_f32_bf16<<<(ND + T - 1) / T, T, 0, stream>>>(x, xb, ND);
    k_f32_bf16<<<(D * D + T - 1) / T, T, 0, stream>>>(W1, W1b, D * D);
    k_f32_bf16<<<(D * D + T - 1) / T, T, 0, stream>>>(W2, W2b, D * D);
    k_f32_bf16<<<(D * D + T - 1) / T, T, 0, stream>>>(Wq, Wqb, D * D);
    k_f32_bf16<<<(D * D + T - 1) / T, T, 0, stream>>>(Wk, Wkb, D * D);
    k_f32_bf16<<<(D * D + T - 1) / T, T, 0, stream>>>(Wv, Wvb, D * D);
    k_f32_bf16<<<(D * D + T - 1) / T, T, 0, stream>>>(Wo, Wob, D * D);
    k_f32_bf16<<<(D * FF + T - 1) / T, T, 0, stream>>>(Wf1, Wf1b, D * FF);
    k_f32_bf16<<<(FF * D + T - 1) / T, T, 0, stream>>>(Wf2, Wf2b, FF * D);

    // ---- local MPNN branch ----
    k_zero<<<(ND + T - 1) / T, T, 0, stream>>>(agg, ND);
    k_edge<<<(E * 128) / T, T, 0, stream>>>(x, ei, ea, agg, E, D);
    k_add_bf16<<<(ND + T - 1) / T, T, 0, stream>>>(x, agg, hinb, ND);
    k_gemm<<<dim3(D / 16, N / 128), T, 0, stream>>>(hinb, W1b, b1, nullptr, nullptr, h1b,
                                                    N, D, D, 1, 1.0f);
    k_gemm<<<dim3(D / 16, N / 128), T, 0, stream>>>(h1b, W2b, b2, x, hl, nullptr,
                                                    N, D, D, 0, 1.0f);

    // ---- global attention branch (operates on x; dense batch == reshape, mask all-true) ----
    k_gemm<<<dim3(D / 16, N / 128), T, 0, stream>>>(xb, Wqb, bq, nullptr, nullptr, qbf,
                                                    N, D, D, 0, qscale);
    k_gemm<<<dim3(D / 16, N / 128), T, 0, stream>>>(xb, Wkb, bk, nullptr, nullptr, kbf,
                                                    N, D, D, 0, 1.0f);
    k_gemm<<<dim3(D / 16, N / 128), T, 0, stream>>>(xb, Wvb, bv, nullptr, nullptr, vbf,
                                                    N, D, D, 0, 1.0f);
    {
        int nwaves = 32 * Hh * (S / 16);                  // B*H*(S/16) = 16384
        k_attn<<<nwaves / 8, T, 0, stream>>>(qbf, kbf, vbf, obf);
    }
    k_gemm<<<dim3(D / 16, N / 128), T, 0, stream>>>(obf, Wob, bo, x, ha, nullptr,
                                                    N, D, D, 0, 1.0f);

    // ---- BN(hl) + BN(ha), combine ----
    k_zero<<<(2 * D + T - 1) / T, T, 0, stream>>>(sums, 2 * D);
    k_col_sums<<<N / RPB, D, 0, stream>>>(hl, sums, RPB, D);
    k_stats_fin<<<(2 * D + T - 1) / T, T, 0, stream>>>(sums, st_l, N, D);
    k_zero<<<(2 * D + T - 1) / T, T, 0, stream>>>(sums, 2 * D);
    k_col_sums<<<N / RPB, D, 0, stream>>>(ha, sums, RPB, D);
    k_stats_fin<<<(2 * D + T - 1) / T, T, 0, stream>>>(sums, st_a, N, D);
    k_combine<<<(ND + T - 1) / T, T, 0, stream>>>(hl, ha, st_l, st_a, g1l, be1l, g1a, be1a,
                                                  hcomb, hbf, ND, D);

    // ---- FFN + final BN ----
    k_gemm<<<dim3(FF / 16, N / 128), T, 0, stream>>>(hbf, Wf1b, bf1, nullptr, nullptr, f1b,
                                                     N, D, FF, 1, 1.0f);
    k_gemm<<<dim3(D / 16, N / 128), T, 0, stream>>>(f1b, Wf2b, bf2, hcomb, f2b_, nullptr,
                                                    N, FF, D, 0, 1.0f);
    k_zero<<<(2 * D + T - 1) / T, T, 0, stream>>>(sums, 2 * D);
    k_col_sums<<<N / RPB, D, 0, stream>>>(f2b_, sums, RPB, D);
    k_stats_fin<<<(2 * D + T - 1) / T, T, 0, stream>>>(sums, st_2, N, D);
    k_bn_out<<<(ND + T - 1) / T, T, 0, stream>>>(f2b_, st_2, g2, be2, (float*)d_out, ND, D);
}
